// T5_MOE_Stack_54846732370209
// MI455X (gfx1250) — matrix operand order repaired and checked by reading, not compiled, not run on hardware
//
#include <hip/hip_runtime.h>
#include <hip/hip_bf16.h>
#include <math.h>

// ---------------------------------------------------------------------------
// CDNA5 (gfx1250) T5+MoE stack.
// GEMMs use v_wmma_f32_16x16x32_bf16 (wave32), everything else VALU.
// ---------------------------------------------------------------------------

typedef __bf16 bf16_t;
typedef __attribute__((ext_vector_type(16))) __bf16 v16bf;
typedef __attribute__((ext_vector_type(8)))  float  v8f;

#define BM 128
#define BN 128
#define BK 32
#define AST 40   // LDS stride (halfs) for A tile rows, keeps 16B alignment
#define BST 40   // LDS stride (halfs) for B tile columns

union Frag { v16bf v; uint4 u[2]; };

// C[M,N] (optionally +=) rowScale * act( A_bf16[M,K] @ W_f32[K,N] )
// act: 0 none, 1 relu, 2 exact gelu.  Cb: optional bf16 copy of result.
__global__ __launch_bounds__(256) void gemm_bf16_wmma(
    const bf16_t* __restrict__ A, const float* __restrict__ W,
    int M, int N, int K,
    float* __restrict__ Cf, bf16_t* __restrict__ Cb,
    const float* __restrict__ rowScale, int rsStride,
    int accumulate, int act)
{
  __shared__ __align__(16) bf16_t Alds[BM * AST];
  __shared__ __align__(16) bf16_t Blds[BN * BST];

  const int tid  = threadIdx.x;
  const int bm   = blockIdx.y * BM;
  const int bn   = blockIdx.x * BN;
  const int wave = tid >> 5;
  const int lane = tid & 31;
  const int wm   = (wave >> 2) * 64;   // 0 / 64
  const int wn   = (wave & 3) * 32;    // 0 / 32 / 64 / 96
  const int l15  = lane & 15;
  const int kh   = lane >> 4;          // lane half selects K sub-range

  v8f acc[4][2];
  #pragma unroll
  for (int i = 0; i < 4; ++i)
    #pragma unroll
    for (int j = 0; j < 2; ++j)
      #pragma unroll
      for (int e = 0; e < 8; ++e) acc[i][j][e] = 0.f;

  // cooperative loader coordinates
  const int ar = tid >> 1;            // A tile row 0..127
  const int ac = (tid & 1) * 16;      // A tile col group {0,16}
  const int br = tid >> 3;            // B tile k-row 0..31
  const int bc = (tid & 7) * 16;      // B tile col group 0..112

  for (int k0 = 0; k0 < K; k0 += BK) {
    // ---- stage A tile (already bf16) : [128][32] ----
    {
      const bf16_t* src = A + (size_t)(bm + ar) * K + k0 + ac;
      uint4 x0 = *(const uint4*)(src);
      uint4 x1 = *(const uint4*)(src + 8);
      *(uint4*)&Alds[ar * AST + ac]     = x0;
      *(uint4*)&Alds[ar * AST + ac + 8] = x1;
    }
    // ---- stage B tile: fp32 -> bf16, transposed into [n][k] ----
    {
      const float* src = W + (size_t)(k0 + br) * N + bn + bc;
      float4 f0 = *(const float4*)(src + 0);
      float4 f1 = *(const float4*)(src + 4);
      float4 f2 = *(const float4*)(src + 8);
      float4 f3 = *(const float4*)(src + 12);
      float fl[16] = {f0.x, f0.y, f0.z, f0.w, f1.x, f1.y, f1.z, f1.w,
                      f2.x, f2.y, f2.z, f2.w, f3.x, f3.y, f3.z, f3.w};
      #pragma unroll
      for (int j = 0; j < 16; ++j)
        Blds[(bc + j) * BST + br] = (bf16_t)fl[j];
    }
    __syncthreads();

    // B fragments: lane holds column n, K in ISA order (K0..7/K16..23, lanes 16-31: +8)
    Frag bfr[2];
    #pragma unroll
    for (int nt = 0; nt < 2; ++nt) {
      const int n = wn + nt * 16 + l15;
      bfr[nt].u[0] = *(const uint4*)&Blds[n * BST + kh * 8];
      bfr[nt].u[1] = *(const uint4*)&Blds[n * BST + 16 + kh * 8];
    }
    #pragma unroll
    for (int mt = 0; mt < 4; ++mt) {
      // A fragment per ISA 16-bit 16x32 layout:
      // lanes 0-15: v0..3 = K0..7, v4..7 = K16..23 ; lanes 16-31: +8
      const int r = wm + mt * 16 + l15;
      Frag af;
      af.u[0] = *(const uint4*)&Alds[r * AST + kh * 8];
      af.u[1] = *(const uint4*)&Alds[r * AST + 16 + kh * 8];
      #pragma unroll
      for (int nt = 0; nt < 2; ++nt)
        acc[mt][nt] = __builtin_amdgcn_wmma_f32_16x16x32_bf16(
            false, af.v, false, bfr[nt].v, (short)0, acc[mt][nt], false, false);
    }
    __syncthreads();
  }

  // ---- epilogue: C layout -> VGPR j: rows j (lanes 0-15) / j+8 (lanes 16-31)
  const int col0 = bn + wn + l15;
  #pragma unroll
  for (int mt = 0; mt < 4; ++mt) {
    #pragma unroll
    for (int nt = 0; nt < 2; ++nt) {
      const int col = col0 + nt * 16;
      #pragma unroll
      for (int j = 0; j < 8; ++j) {
        const int row = bm + wm + mt * 16 + kh * 8 + j;
        float v = acc[mt][nt][j];
        if (act == 1)      v = fmaxf(v, 0.f);
        else if (act == 2) v = 0.5f * v * (1.f + erff(v * 0.7071067811865476f));
        if (rowScale) v *= rowScale[(size_t)row * rsStride];
        const size_t idx = (size_t)row * N + col;
        if (Cf) { if (accumulate) v += Cf[idx]; Cf[idx] = v; }
        if (Cb) Cb[idx] = (bf16_t)v;
      }
    }
  }
}

// ---- T5 RMSNorm: out = w * x * rsqrt(mean(x^2)+eps); fp32 + optional bf16 ---
__global__ __launch_bounds__(256) void rmsnorm_kernel(
    const float* __restrict__ x, const float* __restrict__ w,
    float* __restrict__ outf, bf16_t* __restrict__ outb)
{
  const int row = blockIdx.x;
  const int tid = threadIdx.x;
  __shared__ float red[256];
  const float* xr = x + (size_t)row * 768;
  float x0 = xr[tid], x1 = xr[tid + 256], x2 = xr[tid + 512];
  red[tid] = x0 * x0 + x1 * x1 + x2 * x2;
  __syncthreads();
  for (int off = 128; off > 0; off >>= 1) {
    if (tid < off) red[tid] += red[tid + off];
    __syncthreads();
  }
  const float r = rsqrtf(red[0] * (1.f / 768.f) + 1e-6f);
  const float y0 = w[tid] * x0 * r, y1 = w[tid + 256] * x1 * r, y2 = w[tid + 512] * x2 * r;
  float* of = outf + (size_t)row * 768;
  of[tid] = y0; of[tid + 256] = y1; of[tid + 512] = y2;
  if (outb) {
    bf16_t* ob = outb + (size_t)row * 768;
    ob[tid] = (bf16_t)y0; ob[tid + 256] = (bf16_t)y1; ob[tid + 512] = (bf16_t)y2;
  }
}

// ---- relative-position bucket table for rel = ki-qi in [-511,511] ----------
__global__ void bucket_kernel(int* __restrict__ buckets)
{
  const int i = blockIdx.x * blockDim.x + threadIdx.x;
  if (i >= 1023) return;
  const int rel = i - 511;
  int bkt = (rel > 0) ? 16 : 0;          // nb = NB/2 = 16 (bidirectional)
  const int n = rel < 0 ? -rel : rel;
  if (n < 8) {                            // max_exact = 8
    bkt += n;
  } else {
    int large = 8 + (int)(logf((float)n * 0.125f) * (1.f / logf(16.f)) * 8.f);
    if (large > 15) large = 15;
    bkt += large;
  }
  buckets[i] = bkt;
}

// ---- attention: one block per (qi, h, b); unscaled qk^T + bias, softmax, pV
__global__ __launch_bounds__(256) void attention_kernel(
    const float* __restrict__ Q, const float* __restrict__ Km,
    const float* __restrict__ V, const float* __restrict__ rel_emb,
    const int* __restrict__ buckets, bf16_t* __restrict__ ctx)
{
  const int qi = blockIdx.x, h = blockIdx.y, b = blockIdx.z;
  const int tid = threadIdx.x;
  __shared__ float qs[64];
  __shared__ float sc[512];
  __shared__ float red[256];
  const size_t rowQ = ((size_t)(b * 512 + qi)) * 768 + h * 64;
  if (tid < 64) qs[tid] = Q[rowQ + tid];
  __syncthreads();

  for (int kk = tid; kk < 512; kk += 256) {
    const float* kr = Km + ((size_t)(b * 512 + kk)) * 768 + h * 64;
    float s = 0.f;
    #pragma unroll
    for (int d = 0; d < 64; d += 4) {
      float4 kv = *(const float4*)(kr + d);
      s += qs[d] * kv.x + qs[d + 1] * kv.y + qs[d + 2] * kv.z + qs[d + 3] * kv.w;
    }
    s += rel_emb[buckets[kk - qi + 511] * 12 + h];
    sc[kk] = s;
  }
  __syncthreads();

  red[tid] = fmaxf(sc[tid], sc[tid + 256]);
  __syncthreads();
  for (int off = 128; off > 0; off >>= 1) {
    if (tid < off) red[tid] = fmaxf(red[tid], red[tid + off]);
    __syncthreads();
  }
  const float mx = red[0];
  __syncthreads();
  const float e0 = expf(sc[tid] - mx), e1 = expf(sc[tid + 256] - mx);
  sc[tid] = e0; sc[tid + 256] = e1;
  red[tid] = e0 + e1;
  __syncthreads();
  for (int off = 128; off > 0; off >>= 1) {
    if (tid < off) red[tid] += red[tid + off];
    __syncthreads();
  }
  const float inv = 1.f / red[0];
  __syncthreads();

  const int d = tid & 63, g = tid >> 6;
  float p = 0.f;
  for (int kk = g * 128; kk < g * 128 + 128; ++kk)
    p += sc[kk] * V[((size_t)(b * 512 + kk)) * 768 + h * 64 + d];
  red[tid] = p;
  __syncthreads();
  if (tid < 64) {
    const float v = red[tid] + red[tid + 64] + red[tid + 128] + red[tid + 192];
    ctx[rowQ + tid] = (bf16_t)(v * inv);
  }
}

// ---- NaiveGate: logits = nx @ gw [768x8]; top-2 softmax -> cw[row][8] ------
__global__ __launch_bounds__(256) void gate_kernel(
    const float* __restrict__ nx, const float* __restrict__ gw,
    float* __restrict__ cw)
{
  const int row = blockIdx.x;
  const int tid = threadIdx.x;
  const int e = tid & 7, c = tid >> 3;   // 32 chunks of 24 elems
  __shared__ float red[256];
  __shared__ float logits[8];
  const float* xr = nx + (size_t)row * 768;
  float s = 0.f;
  #pragma unroll
  for (int j = 0; j < 24; ++j) {
    const int d = c * 24 + j;
    s += xr[d] * gw[d * 8 + e];
  }
  red[tid] = s;
  __syncthreads();
  if (tid < 8) {
    float t = 0.f;
    for (int c2 = 0; c2 < 32; ++c2) t += red[c2 * 8 + tid];
    logits[tid] = t;
  }
  __syncthreads();
  if (tid == 0) {
    int i0 = 0; float v0 = logits[0];
    for (int j = 1; j < 8; ++j) if (logits[j] > v0) { v0 = logits[j]; i0 = j; }
    int i1 = -1; float v1 = -3.4e38f;
    for (int j = 0; j < 8; ++j) if (j != i0 && logits[j] > v1) { v1 = logits[j]; i1 = j; }
    const float ex = expf(v1 - v0);
    const float inv = 1.f / (1.f + ex);
    float* out = cw + (size_t)row * 8;
    #pragma unroll
    for (int j = 0; j < 8; ++j) out[j] = 0.f;
    out[i0] = inv;
    out[i1] = ex * inv;
  }
}

// ---------------------------------------------------------------------------
extern "C" void kernel_launch(void* const* d_in, const int* in_sizes, int n_in,
                              void* d_out, int out_size, void* d_ws, size_t ws_size,
                              hipStream_t stream)
{
  (void)in_sizes; (void)n_in; (void)out_size; (void)ws_size;
  const int M = 1024, D = 768, F = 3072;

  const float* x       = (const float*)d_in[0];
  const float* attn_q  = (const float*)d_in[1];
  const float* attn_k  = (const float*)d_in[2];
  const float* attn_v  = (const float*)d_in[3];
  const float* attn_o  = (const float*)d_in[4];
  const float* ffn_wi  = (const float*)d_in[5];
  const float* ffn_wo  = (const float*)d_in[6];
  const float* ln1     = (const float*)d_in[7];
  const float* ln2     = (const float*)d_in[8];
  const float* ln_f    = (const float*)d_in[9];
  const float* rel_emb = (const float*)d_in[10];
  const float* gate_w  = (const float*)d_in[11];
  const float* moe_w1  = (const float*)d_in[12];
  const float* moe_w2  = (const float*)d_in[13];

  char* ws = (char*)d_ws;
  size_t off = 0;
  auto alloc = [&](size_t bytes) -> void* {
    off = (off + 255) & ~(size_t)255;
    void* p = ws + off;
    off += bytes;
    return p;
  };
  float*  h    = (float*) alloc((size_t)M * D * 4);
  float*  nxf  = (float*) alloc((size_t)M * D * 4);
  bf16_t* nxb  = (bf16_t*)alloc((size_t)M * D * 2);
  float*  q    = (float*) alloc((size_t)M * D * 4);
  float*  k    = (float*) alloc((size_t)M * D * 4);
  float*  v    = (float*) alloc((size_t)M * D * 4);
  bf16_t* ctxb = (bf16_t*)alloc((size_t)M * D * 2);
  bf16_t* tb   = (bf16_t*)alloc((size_t)M * F * 2);
  float*  cw   = (float*) alloc((size_t)M * 8 * 4);
  int*    bkts = (int*)   alloc(1023 * 4);

  hipMemcpyAsync(h, x, (size_t)M * D * 4, hipMemcpyDeviceToDevice, stream);
  bucket_kernel<<<4, 256, 0, stream>>>(bkts);

  const dim3 gD(D / 128, M / 128);   // N = 768
  const dim3 gF(F / 128, M / 128);   // N = 3072

  for (int i = 0; i < 4; ++i) {
    // --- self-attention block ---
    rmsnorm_kernel<<<M, 256, 0, stream>>>(h, ln1 + (size_t)i * D, nxf, nxb);
    gemm_bf16_wmma<<<gD, 256, 0, stream>>>(nxb, attn_q + (size_t)i * D * D,
        M, D, D, q, nullptr, nullptr, 0, 0, 0);
    gemm_bf16_wmma<<<gD, 256, 0, stream>>>(nxb, attn_k + (size_t)i * D * D,
        M, D, D, k, nullptr, nullptr, 0, 0, 0);
    gemm_bf16_wmma<<<gD, 256, 0, stream>>>(nxb, attn_v + (size_t)i * D * D,
        M, D, D, v, nullptr, nullptr, 0, 0, 0);
    attention_kernel<<<dim3(512, 12, 2), 256, 0, stream>>>(q, k, v, rel_emb, bkts, ctxb);
    gemm_bf16_wmma<<<gD, 256, 0, stream>>>(ctxb, attn_o + (size_t)i * D * D,
        M, D, D, h, nullptr, nullptr, 0, 1, 0);          // h += ctx @ Wo

    // --- FF block: dense ReLU MLP + additive MoE branch ---
    rmsnorm_kernel<<<M, 256, 0, stream>>>(h, ln2 + (size_t)i * D, nxf, nxb);
    gemm_bf16_wmma<<<gF, 256, 0, stream>>>(nxb, ffn_wi + (size_t)i * D * F,
        M, F, D, nullptr, tb, nullptr, 0, 0, 1);         // t = relu(nx @ Wi)
    gemm_bf16_wmma<<<gD, 256, 0, stream>>>(tb, ffn_wo + (size_t)i * F * D,
        M, D, F, h, nullptr, nullptr, 0, 1, 0);          // h += t @ Wo

    if (i == 1 || i == 3) {
      const int mi = (i == 1) ? 0 : 1;
      gate_kernel<<<M, 256, 0, stream>>>(nxf, gate_w + (size_t)mi * D * 8, cw);
      for (int e = 0; e < 8; ++e) {
        const float* w1 = moe_w1 + ((size_t)(mi * 8 + e)) * D * F;
        const float* w2 = moe_w2 + ((size_t)(mi * 8 + e)) * F * D;
        gemm_bf16_wmma<<<gF, 256, 0, stream>>>(nxb, w1,
            M, F, D, nullptr, tb, nullptr, 0, 0, 2);     // u = gelu(nx @ W1e)
        gemm_bf16_wmma<<<gD, 256, 0, stream>>>(tb, w2,
            M, D, F, h, nullptr, cw + e, 8, 1, 0);       // h += cw[:,e]*(u @ W2e)
      }
    }
  }
  rmsnorm_kernel<<<M, 256, 0, stream>>>(h, ln_f, (float*)d_out, nullptr);
}
